// VQEmbedding_67207648247957
// MI455X (gfx1250) — matrix operand order repaired and checked by reading, not compiled, not run on hardware
//
#include <hip/hip_runtime.h>

typedef __attribute__((ext_vector_type(16))) _Float16 v16h;
typedef __attribute__((ext_vector_type(8)))  float    v8f;

#define K_CODES 512
#define DIM     64

__global__ __launch_bounds__(256) void vq_argmin_gather_kernel(
    const float* __restrict__ z,      // [N, 64] fp32
    const float* __restrict__ emb,    // [512, 64] fp32
    float* __restrict__ out0,         // [N, 64] fp32
    float* __restrict__ out1,         // [N, 64] fp32
    int n_tiles)                      // N / 16
{
    // Codebook staged as f16 for WMMA B-matrix reads + fp32 norms.
    __shared__ __align__(32) _Float16 sEmb[K_CODES * DIM];
    __shared__ float sEn[K_CODES];

    const int tid = threadIdx.x;

    // ---- Stage codebook: fp32 -> f16 in LDS, plus ||e||^2 per code ----
    for (int c = tid; c < K_CODES; c += 256) {
        const float* row = emb + c * DIM;
        float s = 0.0f;
        #pragma unroll
        for (int i = 0; i < DIM; i += 4) {
            float4 f = *(const float4*)(row + i);
            s += f.x * f.x + f.y * f.y + f.z * f.z + f.w * f.w;
            sEmb[c * DIM + i + 0] = (_Float16)f.x;
            sEmb[c * DIM + i + 1] = (_Float16)f.y;
            sEmb[c * DIM + i + 2] = (_Float16)f.z;
            sEmb[c * DIM + i + 3] = (_Float16)f.w;
        }
        sEn[c] = s;
    }
    __syncthreads();

    const int lane  = tid & 31;
    const int wave  = tid >> 5;
    const int gwave = blockIdx.x * 8 + wave;
    const int nwav  = gridDim.x * 8;
    const int hi    = lane >> 4;       // 0 for lanes 0-15, 1 for lanes 16-31
    const int off8  = hi * 8;          // unused; A and B both use off16
    const int off16 = hi * 16;         // B-matrix K-group offset
    const int nsub  = lane & 15;       // N (column) / M (row) sub-index

    for (int rt = gwave; rt < n_tiles; rt += nwav) {
        const int base = rt << 4;

        // ---- Load A panel (16 rows x 64 dims), convert to f16 in WMMA A layout.
        // Lane (hi,m=nsub): chunk h holds K = h*32 + off16 + {0..15}.
        const float* zrow = z + (size_t)(base + nsub) * DIM;
        v16h a0, a1;
        #pragma unroll
        for (int h = 0; h < 2; ++h) {
            const float* p = zrow + h * 32 + off16;
            float4 s0 = *(const float4*)(p + 0);
            float4 s1 = *(const float4*)(p + 4);
            float4 s2 = *(const float4*)(p + 8);
            float4 s3 = *(const float4*)(p + 12);
            v16h a;
            a[0]  = (_Float16)s0.x; a[1]  = (_Float16)s0.y;
            a[2]  = (_Float16)s0.z; a[3]  = (_Float16)s0.w;
            a[4]  = (_Float16)s1.x; a[5]  = (_Float16)s1.y;
            a[6]  = (_Float16)s1.z; a[7]  = (_Float16)s1.w;
            a[8]  = (_Float16)s2.x; a[9]  = (_Float16)s2.y;
            a[10] = (_Float16)s2.z; a[11] = (_Float16)s2.w;
            a[12] = (_Float16)s3.x; a[13] = (_Float16)s3.y;
            a[14] = (_Float16)s3.z; a[15] = (_Float16)s3.w;
            if (h == 0) a0 = a; else a1 = a;
        }

        float bestv[8];
        int   bestc[8];
        #pragma unroll
        for (int v = 0; v < 8; ++v) { bestv[v] = 3.4e38f; bestc[v] = 0; }

        // ---- 32 column tiles of 16 codes: dot via WMMA, fold into running argmin.
        #pragma unroll 4
        for (int ct = 0; ct < 32; ++ct) {
            const int n = (ct << 4) + nsub;   // this lane's codebook column
            // B k order chosen here, same as A, not the ISA table order (32x16 f16): VGPR v holds K=2v,2v+1; lanes<16 K=0..15,
            // lanes>=16 K=16..31 -> 16 contiguous halves per lane.
            const v16h b0 = *(const v16h*)&sEmb[n * DIM + off16];
            const v16h b1 = *(const v16h*)&sEmb[n * DIM + 32 + off16];
            v8f c = {};
            c = __builtin_amdgcn_wmma_f32_16x16x32_f16(false, a0, false, b0,
                                                       (short)0, c, false, false);
            c = __builtin_amdgcn_wmma_f32_16x16x32_f16(false, a1, false, b1,
                                                       (short)0, c, false, false);
            const float en = sEn[n];
            #pragma unroll
            for (int v = 0; v < 8; ++v) {
                // score = ||e||^2 - 2*dot  (||z||^2 constant per row, dropped)
                float sc = fmaf(-2.0f, c[v], en);
                if (sc < bestv[v]) { bestv[v] = sc; bestc[v] = n; }
            }
        }

        // ---- Cross-lane argmin within each 16-lane half (row m = v + 8*hi).
        #pragma unroll
        for (int v = 0; v < 8; ++v) {
            float bv = bestv[v];
            int   bc = bestc[v];
            #pragma unroll
            for (int mask = 1; mask <= 8; mask <<= 1) {
                float ov = __shfl_xor(bv, mask, 32);
                int   oc = __shfl_xor(bc, mask, 32);
                if (ov < bv || (ov == bv && oc < bc)) { bv = ov; bc = oc; }
            }
            bestc[v] = bc;   // now uniform across the 16-lane half
        }

        // ---- Gather winning codebook rows (exact fp32 from L2-hot global)
        // and write both outputs. Row m<8 lives in lanes 0-15, m>=8 in 16-31.
        #pragma unroll
        for (int m = 0; m < 16; ++m) {
            int idx = __shfl(bestc[m & 7], (m < 8) ? 0 : 16, 32);
            const float2 val = ((const float2*)(emb + (size_t)idx * DIM))[lane];
            size_t ro = (size_t)(base + m) * DIM + (size_t)(lane * 2);
            *(float2*)(out0 + ro) = val;
            *(float2*)(out1 + ro) = val;
        }
    }
}

extern "C" void kernel_launch(void* const* d_in, const int* in_sizes, int n_in,
                              void* d_out, int out_size, void* d_ws, size_t ws_size,
                              hipStream_t stream) {
    const float* z   = (const float*)d_in[0];   // [256*2048, 64] fp32
    const float* emb = (const float*)d_in[1];   // [512, 64] fp32
    float* out0 = (float*)d_out;
    const int N = in_sizes[0] / DIM;            // 524288
    float* out1 = out0 + (size_t)N * DIM;
    const int n_tiles = N >> 4;                 // 32768 (N divisible by 16)

    // 512 blocks x 8 waves = 4096 waves -> 8 row-tiles per wave; amortizes the
    // 128KB/block codebook staging while keeping WGPs well occupied.
    dim3 grid(512), block(256);
    vq_argmin_gather_kernel<<<grid, block, 0, stream>>>(z, emb, out0, out1, n_tiles);
}
